// SimCLRLoss_74406013436190
// MI455X (gfx1250) — compile-verified, matrix operand order checked by reading, not run on hardware
//
#include <hip/hip_runtime.h>

typedef _Float16 half_t;
typedef __attribute__((ext_vector_type(16))) _Float16 v16h;
typedef __attribute__((ext_vector_type(8)))  _Float16 v8h;
typedef __attribute__((ext_vector_type(8)))  float    v8f;

#define BROWS 4096
#define DIM   256
#define NROWS 8192      // 2*B
#define TINV  2.0f      // 1 / temperature (0.5)

// ---------------- kernel 0: zero the f32 accumulators in workspace ----------------
__global__ __launch_bounds__(256) void zero_accum(float* __restrict__ p, int n) {
    int i = blockIdx.x * 256 + threadIdx.x;
    if (i < n) p[i] = 0.0f;
}

// ---------------- kernel 1: L2-normalize rows, f32 -> f16 row-major ----------------
// one wave32 per row; lane l holds elements [l*8, l*8+8)
__global__ __launch_bounds__(256) void normalize_rows(const float* __restrict__ z1,
                                                      const float* __restrict__ z2,
                                                      half_t* __restrict__ zh) {
    const int row  = blockIdx.x * 8 + (threadIdx.x >> 5);
    const int lane = threadIdx.x & 31;
    const float* src = (row < BROWS) ? (z1 + (size_t)row * DIM)
                                     : (z2 + (size_t)(row - BROWS) * DIM);
    float4 a = ((const float4*)src)[lane * 2 + 0];
    float4 b = ((const float4*)src)[lane * 2 + 1];
    float s = a.x*a.x + a.y*a.y + a.z*a.z + a.w*a.w
            + b.x*b.x + b.y*b.y + b.z*b.z + b.w*b.w;
    #pragma unroll
    for (int m = 1; m <= 16; m <<= 1) s += __shfl_xor(s, m, 32);
    const float inv = 1.0f / fmaxf(sqrtf(s), 1e-12f);
    v8h o;
    o[0] = (half_t)(a.x * inv); o[1] = (half_t)(a.y * inv);
    o[2] = (half_t)(a.z * inv); o[3] = (half_t)(a.w * inv);
    o[4] = (half_t)(b.x * inv); o[5] = (half_t)(b.y * inv);
    o[6] = (half_t)(b.z * inv); o[7] = (half_t)(b.w * inv);
    *(v8h*)(zh + (size_t)row * DIM + lane * 8) = o;
}

// ---------------- kernel 2: tiled gram matrix + online exp row-reduction ----------------
// block = 256 threads = 8 waves. Wave w owns row-tile (blockIdx.x*8 + w).
// blockIdx.y selects 64 of the 512 column tiles.
// B tiles staged into double-buffered LDS with global_load_async_to_lds_b128 (ASYNCcnt),
// overlapping the next tile's fetch with the current tile's WMMA + exp epilogue.
__global__ __launch_bounds__(256) void gram_loss(const half_t* __restrict__ zh,
                                                 float* __restrict__ neg,
                                                 float* __restrict__ pos) {
    __shared__ __align__(16) half_t lds[2][16 * DIM];   // 2 x 8 KB column-tile stages

    const int lane  = threadIdx.x & 31;
    const int wave  = threadIdx.x >> 5;
    const int l15   = lane & 15;
    const int h     = lane >> 4;                       // lane half (0/1)
    const int mrow0 = (blockIdx.x * 8 + wave) * 16;    // first global row of this wave's tile
    const int ntile0 = blockIdx.y * 64;

    // async stage of one contiguous 8 KB column block into LDS buffer `buf`.
    // NOTE: INST_OFFSET is added to BOTH the global and LDS addresses for async loads,
    // so a single offset:4096 instruction moves the second half.
    auto stage = [&](int nt, int buf) {
        const half_t* src = zh + (size_t)((ntile0 + nt) * 16) * DIM;
        uint64_t g = (uint64_t)(uintptr_t)src + (uint64_t)threadIdx.x * 16u;
        uint32_t l = (uint32_t)(uintptr_t)(&lds[buf][0]) + (uint32_t)threadIdx.x * 16u;
        asm volatile("global_load_async_to_lds_b128 %0, %1, off"
                     :: "v"(l), "v"(g) : "memory");
        asm volatile("global_load_async_to_lds_b128 %0, %1, off offset:4096"
                     :: "v"(l), "v"(g) : "memory");
    };

    // A fragments: 16 rows x K=256 of f16 held in registers for the whole kernel.
    // 16-bit A layout: lane half h holds K {kk*32 + h*8 ..+7} in halves 0..7
    // and K {kk*32 + 16 + h*8 ..+7} in halves 8..15.
    v16h afrag[8];
    {
        const half_t* arow = zh + (size_t)(mrow0 + l15) * DIM;
        #pragma unroll
        for (int kk = 0; kk < 8; ++kk) {
            ((v8h*)&afrag[kk])[0] = *(const v8h*)(arow + kk * 32 + h * 8);
            ((v8h*)&afrag[kk])[1] = *(const v8h*)(arow + kk * 32 + 16 + h * 8);
        }
    }

    float acc[8];   // per-lane partial exp row-sums (diag excluded)
    float pacc[8];  // per-lane positive-pair sim capture
    #pragma unroll
    for (int r = 0; r < 8; ++r) { acc[r] = 0.0f; pacc[r] = 0.0f; }

    stage(0, 0);    // prologue: tile 0 in flight

    for (int nt = 0; nt < 64; ++nt) {
        const int buf = nt & 1;
        asm volatile("s_wait_asynccnt 0x0" ::: "memory");  // tile nt landed in LDS
        __syncthreads();
        if (nt + 1 < 64) stage(nt + 1, buf ^ 1);           // overlap next fetch with compute

        const half_t* brow = &lds[buf][0] + l15 * DIM;     // X*X^T: B frags use the A pattern
        v16h bfr[2];
        #define LOAD_B(dst, kk)                                                   \
            do {                                                                  \
                ((v8h*)&(dst))[0] = *(const v8h*)(brow + (kk) * 32 + h * 8);      \
                ((v8h*)&(dst))[1] = *(const v8h*)(brow + (kk) * 32 + 16 + h * 8); \
            } while (0)

        LOAD_B(bfr[0], 0);
        v8f c = {};
        #pragma unroll
        for (int kk = 0; kk < 8; ++kk) {
            if (kk + 1 < 8) LOAD_B(bfr[(kk + 1) & 1], kk + 1);   // pipeline: fetch ahead
            c = __builtin_amdgcn_wmma_f32_16x16x32_f16(
                    false, afrag[kk], false, bfr[kk & 1], (short)0, c, false, false);
        }
        #undef LOAD_B

        // C layout: VGPR r, lanes 0-15 -> (M=r, N=lane); lanes 16-31 -> (M=8+r, N=lane-16)
        const int col = ntile0 * 16 + nt * 16 + l15;
        #pragma unroll
        for (int r = 0; r < 8; ++r) {
            const int row = mrow0 + h * 8 + r;
            const float s = c[r];
            const float e = __expf(TINV * s);
            acc[r]  += (col == row) ? 0.0f : e;                           // rowsum minus diag
            pacc[r] += (col == ((row + BROWS) & (NROWS - 1))) ? s : 0.0f; // positive pair sim
        }
    }

    // reduce over the 16 lanes of each half, then atomically accumulate per-row
    #pragma unroll
    for (int r = 0; r < 8; ++r) {
        float a = acc[r], p = pacc[r];
        #pragma unroll
        for (int m = 1; m <= 8; m <<= 1) {
            a += __shfl_xor(a, m, 32);
            p += __shfl_xor(p, m, 32);
        }
        if (l15 == 0) {
            const int row = mrow0 + h * 8 + r;
            atomicAdd(&neg[row], a);
            atomicAdd(&pos[row], p);
        }
    }
}

// ---------------- kernel 3: finalize  mean(log(neg) - sim_pos/T) ----------------
__global__ __launch_bounds__(256) void finalize(const float* __restrict__ neg,
                                                const float* __restrict__ pos,
                                                float* __restrict__ out) {
    __shared__ float red[256];
    float s = 0.0f;
    for (int j = threadIdx.x; j < NROWS; j += 256)
        s += logf(neg[j]) - TINV * pos[j];
    red[threadIdx.x] = s;
    __syncthreads();
    for (int off = 128; off > 0; off >>= 1) {
        if (threadIdx.x < off) red[threadIdx.x] += red[threadIdx.x + off];
        __syncthreads();
    }
    if (threadIdx.x == 0) out[0] = red[0] / (float)NROWS;
}

extern "C" void kernel_launch(void* const* d_in, const int* in_sizes, int n_in,
                              void* d_out, int out_size, void* d_ws, size_t ws_size,
                              hipStream_t stream) {
    const float* z1 = (const float*)d_in[0];
    const float* z2 = (const float*)d_in[1];
    float* out = (float*)d_out;

    // workspace layout: [ zh f16 8192x256 = 4 MB ][ neg f32 8192 ][ pos f32 8192 ]
    half_t* zh  = (half_t*)d_ws;
    float*  neg = (float*)((char*)d_ws + (size_t)NROWS * DIM * sizeof(half_t));
    float*  pos = neg + NROWS;

    zero_accum    <<<(2 * NROWS + 255) / 256, 256, 0, stream>>>(neg, 2 * NROWS);
    normalize_rows<<<NROWS / 8,               256, 0, stream>>>(z1, z2, zh);
    dim3 grid(NROWS / (16 * 8), 8);   // 64 row-blocks x 8 column splits
    gram_loss     <<<grid,                    256, 0, stream>>>(zh, neg, pos);
    finalize      <<<1,                       256, 0, stream>>>(neg, pos, out);
}
